// RNN_33079838114438
// MI455X (gfx1250) — matrix operand order repaired and checked by reading, not compiled, not run on hardware
//
#include <hip/hip_runtime.h>
#include <hip/hip_bf16.h>

typedef __attribute__((ext_vector_type(16))) __bf16        v16bf;
typedef __attribute__((ext_vector_type(8)))  float         v8f;
typedef __attribute__((ext_vector_type(4)))  unsigned int  u32x4;

#define B_    128
#define T_    1024
#define H_    512
#define NWG   32     // workgroups; each owns 16 hidden units
#define UNITS 16
#define KP    520    // padded K stride in bf16 elems (1040 B, 16B multiple)

// gfx1250 has V_TANH_F32 (TRANS op). Use the builtin when the toolchain
// exposes it; fall back to libm tanhf otherwise.
#if __has_builtin(__builtin_amdgcn_tanhf)
#define TANHF(v) __builtin_amdgcn_tanhf(v)
#elif __has_builtin(__builtin_amdgcn_tanh_f32)
#define TANHF(v) __builtin_amdgcn_tanh_f32(v)
#else
#define TANHF(v) tanhf(v)
#endif

// sigmoid via tanh: one TRANS op instead of exp + rcp
__device__ __forceinline__ float sigf(float v) {
  return 0.5f * TANHF(0.5f * v) + 0.5f;
}

// Low 32 bits of a generic pointer to __shared__ data == LDS byte offset
// (FLAT shared-aperture mapping keeps the offset in the low 32 bits).
__device__ __forceinline__ unsigned lds_off(const void* p) {
  return (unsigned)(unsigned long long)p;
}

// Re-initializes all cross-call state every launch (graph-replay safe):
// out <- bl (readout bias), both h ping-pong buffers <- 0, barrier words <- 0.
__global__ void __launch_bounds__(256) lstm_init(float* __restrict__ out,
                                                 __bf16* __restrict__ hbuf,
                                                 unsigned* __restrict__ sync,
                                                 const float* __restrict__ bl) {
  int i = blockIdx.x * blockDim.x + threadIdx.x;
  if (i < 2) sync[i] = 0u;
  if (i < B_ * T_) out[i] = bl[0];
  if (i < 2 * B_ * H_) hbuf[i] = (__bf16)0.0f;
}

__global__ void __launch_bounds__(256, 1) lstm_persistent(
    const float* __restrict__ x,     // (B,T,1)
    const float* __restrict__ Wx,    // (1,4H)
    const float* __restrict__ Wh,    // (H,4H)
    const float* __restrict__ bias,  // (4H)
    const float* __restrict__ Wl,    // (H,1)
    float* __restrict__ out,         // (B,T)
    __bf16* __restrict__ hbuf,       // 2 * (B,H) ping-pong
    unsigned* __restrict__ sync)     // [0]=counter [1]=generation
{
  // Per-wave A staging (16 rows of h) + per-WG Wh^T slice.  ~195 KB static LDS.
  __shared__ __align__(16) __bf16 hS[8][16 * KP];
  __shared__ __align__(16) __bf16 wS[4 * UNITS * KP];

  const int tid  = threadIdx.x;
  const int w    = tid >> 5;    // wave id 0..7 (wave32)
  const int lane = tid & 31;
  const int half = lane >> 4;   // 0: lanes 0-15, 1: lanes 16-31
  const int lm   = lane & 15;
  const int j0   = blockIdx.x * UNITS;

  // ---- one-time: stage Wh^T (bf16, column-major-in-K) into LDS ----
  // wS[c*KP + k] = Wh[k][gate*H + j0 + u],  c = gate*16 + u
  for (int idx = tid; idx < 4 * UNITS * H_; idx += 256) {
    int k = idx >> 6;        // 0..511
    int c = idx & 63;        // 0..63
    int gate = c >> 4, u = c & 15;
    wS[c * KP + k] = (__bf16)Wh[k * (4 * H_) + gate * H_ + j0 + u];
  }
  __syncthreads();

  // per-lane constants: this lane's gate column is n = g*H + j0 + lm
  float wxc[4], bc[4];
#pragma unroll
  for (int g = 0; g < 4; ++g) {
    int n = g * H_ + j0 + lm;
    wxc[g] = Wx[n];
    bc[g]  = bias[n];
  }
  const float wlv = Wl[j0 + lm];

  // persistent cell state: lane holds c for rows (16w + r + 8*half), unit (j0+lm)
  float cst[8];
#pragma unroll
  for (int r = 0; r < 8; ++r) cst[r] = 0.0f;

  const int rowBase = w * 16;
  const unsigned hSw = lds_off(&hS[w][0]);
  unsigned* cnt = sync;
  unsigned* gen = sync + 1;

  for (int t = 0; t < T_; ++t) {
    const __bf16* __restrict__ hR = hbuf + (size_t)(t & 1) * (B_ * H_);
    __bf16* __restrict__ hW       = hbuf + (size_t)((t + 1) & 1) * (B_ * H_);

    // ---- async-stage this wave's 16 rows of h into LDS ----
    // GLOBAL_LOAD_ASYNC_TO_LDS_B128, GVS form: per-lane LDS addr (VGPR),
    // per-lane byte offset (VGPR), uniform base (SGPR pair). ASYNCcnt-tracked.
    for (int i = lane; i < 1024; i += 32) {   // 1024 chunks of 8 bf16 (16B)
      int row = i >> 6;                       // 64 chunks per row
      int ck  = (i & 63) << 3;
      unsigned l = hSw + (unsigned)(row * KP + ck) * 2u;
      unsigned g = (unsigned)(((rowBase + row) * H_ + ck) * 2);
      asm volatile("global_load_async_to_lds_b128 %0, %1, %2"
                   :
                   : "v"(l), "v"(g), "s"(hR)
                   : "memory");
    }

    // x_t contribution (I==1 -> outer product) folded into the accumulator init
    float xv[8];
#pragma unroll
    for (int r = 0; r < 8; ++r)
      xv[r] = x[(rowBase + r + half * 8) * T_ + t];
    if (t + 1 < T_)
      __builtin_prefetch(&x[(rowBase + half * 8) * T_ + t + 1], 0, 3);

    v8f acc[4];
#pragma unroll
    for (int g = 0; g < 4; ++g)
#pragma unroll
      for (int r = 0; r < 8; ++r)
        acc[g][r] = __builtin_fmaf(xv[r], wxc[g], bc[g]);

    // wait for the async tile before touching hS
    asm volatile("s_wait_asynccnt 0x0" ::: "memory");

    // ---- K loop: h(16x512) x WhT -> 4 gate tiles, A reused across gates ----
    const __bf16* aRow = &hS[w][lm * KP];
#pragma unroll 4
    for (int kt = 0; kt < H_; kt += 32) {
      // A 16x32 bf16: lanes 0-15 carry K {0..7,16..23}, lanes 16-31 {8..15,24..31}
      union { u32x4 q[2]; v16bf v; } a;
      a.q[0] = *(const u32x4*)(aRow + kt + half * 8);
      a.q[1] = *(const u32x4*)(aRow + kt + 16 + half * 8);
#pragma unroll
      for (int g = 0; g < 4; ++g) {
        // B 32x16 bf16: lanes 0-15 carry K {0..7,16..23}, lanes 16-31 {8..15,24..31}
        union { u32x4 q[2]; v16bf v; } bv;
        const __bf16* bp = &wS[(g * 16 + lm) * KP + kt];
        bv.q[0] = *(const u32x4*)(bp + half * 8);
        bv.q[1] = *(const u32x4*)(bp + 16 + half * 8);
        acc[g] = __builtin_amdgcn_wmma_f32_16x16x32_bf16(
            false, a.v, false, bv.v, (short)0, acc[g], false, false);
      }
    }

    // ---- fused LSTM cell (register-local per lane) + readout partials ----
    float pr[8];
#pragma unroll
    for (int r = 0; r < 8; ++r) {
      float iv = sigf(acc[0][r]);
      float fv = sigf(acc[1][r]);
      float gv = TANHF(acc[2][r]);
      float ov = sigf(acc[3][r]);
      cst[r] = fv * cst[r] + iv * gv;
      float hv = ov * TANHF(cst[r]);
      hW[(rowBase + r + half * 8) * H_ + j0 + lm] = (__bf16)hv;
      pr[r] = hv * wlv;
    }
    // reduce over the 16 units (lanes within each 16-lane half)
#pragma unroll
    for (int m = 8; m >= 1; m >>= 1)
#pragma unroll
      for (int r = 0; r < 8; ++r)
        pr[r] += __shfl_xor(pr[r], m, 32);
    if (lm == 0) {
#pragma unroll
      for (int r = 0; r < 8; ++r)
        atomicAdd(&out[(rowBase + r + half * 8) * T_ + t], pr[r]);
    }

    // ---- grid-wide barrier (sense-counting, device scope) ----
    __threadfence();
    __syncthreads();
    if (tid == 0) {
      unsigned arrived = __hip_atomic_fetch_add(cnt, 1u, __ATOMIC_ACQ_REL,
                                                __HIP_MEMORY_SCOPE_AGENT);
      if (arrived == NWG - 1) {
        __hip_atomic_store(cnt, 0u, __ATOMIC_RELAXED, __HIP_MEMORY_SCOPE_AGENT);
        __hip_atomic_fetch_add(gen, 1u, __ATOMIC_RELEASE,
                               __HIP_MEMORY_SCOPE_AGENT);
      } else {
        while (__hip_atomic_load(gen, __ATOMIC_ACQUIRE,
                                 __HIP_MEMORY_SCOPE_AGENT) < (unsigned)(t + 1))
          __builtin_amdgcn_s_sleep(2);
      }
    }
    __syncthreads();
  }
}

extern "C" void kernel_launch(void* const* d_in, const int* in_sizes, int n_in,
                              void* d_out, int out_size, void* d_ws,
                              size_t ws_size, hipStream_t stream) {
  const float* x    = (const float*)d_in[0];
  const float* Wx   = (const float*)d_in[1];
  const float* Wh   = (const float*)d_in[2];
  const float* bias = (const float*)d_in[3];
  const float* Wl   = (const float*)d_in[4];
  const float* bl   = (const float*)d_in[5];
  // d_in[6] = future (0), unused

  float*    out  = (float*)d_out;
  unsigned* sync = (unsigned*)d_ws;                      // 2 words + pad
  __bf16*   hbuf = (__bf16*)((char*)d_ws + 256);         // 2 * B*H bf16

  int initN = B_ * T_;  // == 2*B_*H_ as well
  lstm_init<<<(initN + 255) / 256, 256, 0, stream>>>(out, hbuf, sync, bl);
  lstm_persistent<<<NWG, 256, 0, stream>>>(x, Wx, Wh, bias, Wl, out, hbuf,
                                           sync);
}
